// DGCNN_simple_53601191854513
// MI455X (gfx1250) — matrix operand order repaired and checked by reading, not compiled, not run on hardware
//
#include <hip/hip_runtime.h>
#include <stdint.h>

// ---------------------------------------------------------------------------
// DGCNN on gfx1250 (MI455X): WMMA f16 GEMMs for Gram matrices + pointwise
// convs, scalar top-k, BN(train) via atomically-reduced channel stats.
// ---------------------------------------------------------------------------

typedef _Float16 h16;
typedef __attribute__((ext_vector_type(16))) _Float16 v16h;
typedef __attribute__((ext_vector_type(8)))  _Float16 v8h;
typedef __attribute__((ext_vector_type(8)))  float    v8f;

#define B_   4
#define N_   4096
#define KNN  20
#define HCH  64
#define R_   (B_ * N_ * KNN)      // 327680 rows, exactly 20480 * 16
#define MT_  (R_ / 16)            // 20480 M-tiles
#define SLOPE 0.2f
#define EPS   1e-5f

// ---------------- small helpers ----------------

__device__ __forceinline__ void topk_insert(float* bv, int* bi, float v, int m) {
  int p = KNN - 1;
  while (p > 0 && bv[p - 1] < v) { bv[p] = bv[p - 1]; bi[p] = bi[p - 1]; --p; }
  bv[p] = v; bi[p] = m;
}

__device__ __forceinline__ v16h make_a_frag(const h16* base, int hi) {
  // A 16x32 f16: lane (hi half) holds K = {8*hi..8*hi+7, 16+8*hi..23+8*hi}
  const v8h a0 = *(const v8h*)(base + hi * 8);
  const v8h a1 = *(const v8h*)(base + hi * 8 + 16);
  v16h a;
#pragma unroll
  for (int i = 0; i < 8; ++i) { a[i] = a0[i]; a[i + 8] = a1[i]; }
  return a;
}

__device__ __forceinline__ v16h make_b_frag(const h16* base, int hi) {
  // B 32x16 f16: lane (hi half) holds K = {8*hi..8*hi+7, 16+8*hi..23+8*hi}
  const v8h b0 = *(const v8h*)(base + hi * 8);
  const v8h b1 = *(const v8h*)(base + hi * 8 + 16);
  v16h b;
#pragma unroll
  for (int i = 0; i < 8; ++i) { b[i] = b0[i]; b[i + 8] = b1[i]; }
  return b;
}

// ---------------- weights: f32 (Cout x Cin) -> f16 (Cout x Cin_pad) ----------

__global__ void k_cvt_w(const float* __restrict__ w, h16* __restrict__ out,
                        int cin, int cin_pad) {
  int t = blockIdx.x * blockDim.x + threadIdx.x;
  if (t >= HCH * cin_pad) return;
  int o = t / cin_pad, c = t % cin_pad;
  out[t] = (c < cin) ? (h16)w[o * cin + c] : (h16)0.0f;
}

__global__ void k_zero(float* __restrict__ p, int n) {
  int t = blockIdx.x * blockDim.x + threadIdx.x;
  if (t < n) p[t] = 0.0f;
}

// ---------------- kNN for C=3 (block 1), direct scalar scan -----------------

__global__ void k_knn3(const float* __restrict__ x, int* __restrict__ idx) {
  int t = blockIdx.x * blockDim.x + threadIdx.x;   // b*N + n
  if (t >= B_ * N_) return;
  int b = t / N_, n = t % N_;
  const float* xb = x + (size_t)b * N_ * 3;
  float cx = xb[n * 3 + 0], cy = xb[n * 3 + 1], cz = xb[n * 3 + 2];
  float bv[KNN]; int bi[KNN];
#pragma unroll
  for (int i = 0; i < KNN; ++i) { bv[i] = -3e38f; bi[i] = 0; }
  for (int m = 0; m < N_; ++m) {
    float dx = xb[m * 3 + 0] - cx;
    float dy = xb[m * 3 + 1] - cy;
    float dz = xb[m * 3 + 2] - cz;
    float v = -(dx * dx + dy * dy + dz * dz);
    if (v > bv[KNN - 1]) topk_insert(bv, bi, v, m);
  }
  int* op = idx + (size_t)t * KNN;
  for (int i = 0; i < KNN; ++i) op[i] = bi[i];
}

// ---------------- Gram tile (one batch): dist = 2*x.xT - |x|^2 - |x|^2 ------
// grid (N/16, N/16), block 32 (one wave). K = 64 -> two WMMA f16 steps.

__global__ void k_gram(const h16* __restrict__ xh, const float* __restrict__ xx,
                       float* __restrict__ dist) {
  int lane = threadIdx.x;
  int l16 = lane & 15, hi = lane >> 4;
  int arow = blockIdx.y * 16 + l16;
  int bcol = blockIdx.x * 16 + l16;
  v8f c = {};
#pragma unroll
  for (int kc = 0; kc < 64; kc += 32) {
    v16h a = make_a_frag(xh + (size_t)arow * 64 + kc, hi);
    v16h b = make_b_frag(xh + (size_t)bcol * 64 + kc, hi);
    c = __builtin_amdgcn_wmma_f32_16x16x32_f16(false, a, false, b, (short)0, c,
                                               false, false);
  }
  float xc = xx[bcol];
#pragma unroll
  for (int r = 0; r < 8; ++r) {
    int row = blockIdx.y * 16 + r + 8 * hi;
    dist[(size_t)row * N_ + bcol] = 2.0f * c[r] - xx[row] - xc;
  }
}

// ---------------- top-k over one distance row -------------------------------

__global__ void k_topk(const float* __restrict__ dist, int* __restrict__ idx) {
  int n = blockIdx.x * blockDim.x + threadIdx.x;
  if (n >= N_) return;
  const float* dp = dist + (size_t)n * N_;
  float bv[KNN]; int bi[KNN];
#pragma unroll
  for (int i = 0; i < KNN; ++i) { bv[i] = -3e38f; bi[i] = 0; }
  for (int m = 0; m < N_; ++m) {
    float v = dp[m];
    if (v > bv[KNN - 1]) topk_insert(bv, bi, v, m);
  }
  int* op = idx + (size_t)n * KNN;
  for (int i = 0; i < KNN; ++i) op[i] = bi[i];
}

// ---------------- edge features ---------------------------------------------

__global__ void k_edge6(const float* __restrict__ x, const int* __restrict__ idx,
                        h16* __restrict__ aout) {
  size_t t = (size_t)blockIdx.x * blockDim.x + threadIdx.x;
  if (t >= (size_t)R_ * 32) return;
  size_t r = t >> 5; int c = (int)(t & 31);
  size_t bn = r / KNN;
  int b = (int)(bn / N_);
  float v = 0.0f;
  if (c < 6) {
    int j = idx[r];
    size_t jrow = (size_t)b * N_ + j;
    if (c < 3) v = x[jrow * 3 + c] - x[bn * 3 + c];
    else       v = x[bn * 3 + (c - 3)];
  }
  aout[t] = (h16)v;
}

__global__ void k_edge128(const h16* __restrict__ xh, const int* __restrict__ idx,
                          h16* __restrict__ aout) {
  size_t t = (size_t)blockIdx.x * blockDim.x + threadIdx.x;
  if (t >= (size_t)R_ * 128) return;
  size_t r = t >> 7; int c = (int)(t & 127);
  size_t bn = r / KNN;
  int b = (int)(bn / N_);
  float v;
  if (c < 64) {
    int j = idx[r];
    size_t jrow = (size_t)b * N_ + j;
    v = (float)xh[jrow * 64 + c] - (float)xh[bn * 64 + c];
  } else {
    v = (float)xh[bn * 64 + (c - 64)];
  }
  aout[t] = (h16)v;
}

// ---------------- pointwise conv: Y[R x 64] = A[R x Cin] * W^T --------------
// block = 128 threads (4 waves); each block owns a 16-row tile, wave w owns
// output columns [16w, 16w+16). Also accumulates per-channel sum/sumsq.

__global__ void k_conv(const h16* __restrict__ act, const h16* __restrict__ wgt,
                       float* __restrict__ y, float* __restrict__ gsum,
                       float* __restrict__ gsq, int cin_pad) {
  __shared__ __align__(16) h16 sA[16 * 128];
  __shared__ float s_sum[64], s_sq[64];
  int tid = threadIdx.x;
  int wv = tid >> 5, lane = tid & 31;
  int l16 = lane & 15, hi = lane >> 4;
  if (tid < 64) { s_sum[tid] = 0.0f; s_sq[tid] = 0.0f; }

  size_t m = blockIdx.x;
  const uint4* gsrc = (const uint4*)(act + m * 16 * cin_pad);
  uint4* ldst = (uint4*)sA;
  int nvec = 16 * cin_pad / 8;
  for (int i = tid; i < nvec; i += 128) ldst[i] = gsrc[i];
  __syncthreads();

  int ncol = wv * 16 + l16;
  v8f c = {};
  for (int kc = 0; kc < cin_pad; kc += 32) {
    v16h a = make_a_frag(sA + (size_t)l16 * cin_pad + kc, hi);
    v16h b = make_b_frag(wgt + (size_t)ncol * cin_pad + kc, hi);
    c = __builtin_amdgcn_wmma_f32_16x16x32_f16(false, a, false, b, (short)0, c,
                                               false, false);
  }

  float ls = 0.0f, lq = 0.0f;
#pragma unroll
  for (int r = 0; r < 8; ++r) {
    float v = c[r];
    y[(m * 16 + r + 8 * hi) * 64 + ncol] = v;
    ls += v; lq += v * v;
  }
  atomicAdd(&s_sum[ncol], ls);
  atomicAdd(&s_sq[ncol], lq);
  __syncthreads();
  if (tid < 64) {
    atomicAdd(&gsum[tid], s_sum[tid]);
    atomicAdd(&gsq[tid], s_sq[tid]);
  }
}

// ---------------- BN stats finalize -----------------------------------------

__global__ void k_stats(const float* __restrict__ gsum, const float* __restrict__ gsq,
                        const float* __restrict__ gamma, const float* __restrict__ beta,
                        float* __restrict__ scale, float* __restrict__ shift,
                        float invcnt) {
  int c = threadIdx.x;
  if (c >= 64) return;
  float mean = gsum[c] * invcnt;
  float var  = gsq[c] * invcnt - mean * mean;
  float sc   = gamma[c] * rsqrtf(var + EPS);
  scale[c] = sc;
  shift[c] = beta[c] - mean * sc;
}

// ---------------- BN + LeakyReLU (to f16 activations) -----------------------

__global__ void k_bn_act(const float* __restrict__ y, const float* __restrict__ scale,
                         const float* __restrict__ shift, h16* __restrict__ aout) {
  size_t t = (size_t)blockIdx.x * blockDim.x + threadIdx.x;
  if (t >= (size_t)R_ * 64) return;
  int c = (int)(t & 63);
  float v = y[t] * scale[c] + shift[c];
  v = (v >= 0.0f) ? v : SLOPE * v;
  aout[t] = (h16)v;
}

// ---------------- BN + LeakyReLU + max over k -------------------------------
// block per (b,n), 64 threads (one per channel). Optionally emits the f16
// feature row + squared norm for the next kNN stage, and/or f32 output.

__global__ void k_bn_act_max(const float* __restrict__ y,
                             const float* __restrict__ scale,
                             const float* __restrict__ shift,
                             h16* __restrict__ xh, float* __restrict__ xx,
                             float* __restrict__ fout) {
  __shared__ float s_nrm[64];
  int c = threadIdx.x;
  size_t row = blockIdx.x;                 // b*N + n
  const float* yp = y + row * KNN * 64 + c;
  float sc = scale[c], sh = shift[c];
  float mx = -3e38f;
#pragma unroll 4
  for (int kk = 0; kk < KNN; ++kk) {
    float v = yp[(size_t)kk * 64] * sc + sh;
    v = (v >= 0.0f) ? v : SLOPE * v;
    mx = fmaxf(mx, v);
  }
  if (fout) fout[row * 64 + c] = mx;
  if (xh) {
    xh[row * 64 + c] = (h16)mx;
    s_nrm[c] = mx * mx;
    __syncthreads();
    if (c == 0) {
      float s = 0.0f;
      for (int i = 0; i < 64; ++i) s += s_nrm[i];
      xx[row] = s;
    }
  }
}

// ---------------------------------------------------------------------------
// Host orchestration
// ---------------------------------------------------------------------------

extern "C" void kernel_launch(void* const* d_in, const int* in_sizes, int n_in,
                              void* d_out, int out_size, void* d_ws, size_t ws_size,
                              hipStream_t stream) {
  (void)in_sizes; (void)n_in; (void)out_size; (void)ws_size;

  const float* x     = (const float*)d_in[0];
  const float* gamma = (const float*)d_in[10];
  const float* beta  = (const float*)d_in[11];
  float* out         = (float*)d_out;

  // workspace layout (256B aligned slabs)
  char* ws = (char*)d_ws;
  size_t off = 0;
  auto slab = [&](size_t bytes) {
    size_t r = off;
    off = (off + bytes + 255) & ~(size_t)255;
    return r;
  };
  float* dist  = (float*)(ws + slab((size_t)N_ * N_ * 4));          // 67 MB (per-batch reuse)
  h16*   act   = (h16*)  (ws + slab((size_t)R_ * 128 * 2));         // 84 MB
  float* ybuf  = (float*)(ws + slab((size_t)R_ * 64 * 4));          // 84 MB
  h16*   xh1   = (h16*)  (ws + slab((size_t)B_ * N_ * 64 * 2));     // 2 MB
  h16*   xh2   = (h16*)  (ws + slab((size_t)B_ * N_ * 64 * 2));     // 2 MB
  float* xx    = (float*)(ws + slab((size_t)B_ * N_ * 4));
  int*   idx   = (int*)  (ws + slab((size_t)B_ * N_ * KNN * 4));
  h16*   wh    = (h16*)  (ws + slab((size_t)9 * 64 * 128 * 2));
  float* gsum  = (float*)(ws + slab(64 * 4));
  float* gsq   = (float*)(ws + slab(64 * 4));
  float* scale = (float*)(ws + slab(64 * 4));
  float* shift = (float*)(ws + slab(64 * 4));

  const int layer_cin[9]     = {6, 64, 64, 128, 64, 64, 128, 64, 64};
  const int layer_cin_pad[9] = {32, 64, 64, 128, 64, 64, 128, 64, 64};

  // convert all 9 weight matrices to padded f16
  for (int l = 0; l < 9; ++l) {
    int cp = layer_cin_pad[l];
    int n = HCH * cp;
    k_cvt_w<<<(n + 255) / 256, 256, 0, stream>>>((const float*)d_in[1 + l],
                                                 wh + (size_t)l * 64 * 128,
                                                 layer_cin[l], cp);
  }

  const float invcnt = 1.0f / (float)R_;

  auto conv_layer = [&](int l) {
    int cp = layer_cin_pad[l];
    k_zero<<<1, 128, 0, stream>>>(gsum, 64);
    k_zero<<<1, 128, 0, stream>>>(gsq, 64);
    k_conv<<<MT_, 128, 0, stream>>>(act, wh + (size_t)l * 64 * 128, ybuf, gsum,
                                    gsq, cp);
    k_stats<<<1, 64, 0, stream>>>(gsum, gsq, gamma + 64 * l, beta + 64 * l,
                                  scale, shift, invcnt);
  };
  auto bn_act = [&]() {
    size_t tot = (size_t)R_ * 64;
    k_bn_act<<<(unsigned)((tot + 255) / 256), 256, 0, stream>>>(ybuf, scale,
                                                                shift, act);
  };
  auto knn64 = [&](const h16* feat) {
    for (int b = 0; b < B_; ++b) {
      k_gram<<<dim3(N_ / 16, N_ / 16), 32, 0, stream>>>(
          feat + (size_t)b * N_ * 64, xx + (size_t)b * N_, dist);
      k_topk<<<N_ / 128, 128, 0, stream>>>(dist, idx + (size_t)b * N_ * KNN);
    }
  };

  // ---------------- block 1 (C = 3) ----------------
  k_knn3<<<(B_ * N_) / 128, 128, 0, stream>>>(x, idx);
  {
    size_t tot = (size_t)R_ * 32;
    k_edge6<<<(unsigned)((tot + 255) / 256), 256, 0, stream>>>(x, idx, act);
  }
  conv_layer(0); bn_act();
  conv_layer(1); bn_act();
  conv_layer(2);
  k_bn_act_max<<<B_ * N_, 64, 0, stream>>>(ybuf, scale, shift, xh1, xx, nullptr);

  // ---------------- block 2 (C = 64) ----------------
  knn64(xh1);
  {
    size_t tot = (size_t)R_ * 128;
    k_edge128<<<(unsigned)((tot + 255) / 256), 256, 0, stream>>>(xh1, idx, act);
  }
  conv_layer(3); bn_act();
  conv_layer(4); bn_act();
  conv_layer(5);
  k_bn_act_max<<<B_ * N_, 64, 0, stream>>>(ybuf, scale, shift, xh2, xx, nullptr);

  // ---------------- block 3 (C = 64) ----------------
  knn64(xh2);
  {
    size_t tot = (size_t)R_ * 128;
    k_edge128<<<(unsigned)((tot + 255) / 256), 256, 0, stream>>>(xh2, idx, act);
  }
  conv_layer(6); bn_act();
  conv_layer(7); bn_act();
  conv_layer(8);
  k_bn_act_max<<<B_ * N_, 64, 0, stream>>>(ybuf, scale, shift, nullptr, nullptr,
                                           out);
}
